// RelaxQuantizer_17815524344476
// MI455X (gfx1250) — matrix operand order repaired and checked by reading, not compiled, not run on hardware
//
#include <hip/hip_runtime.h>
#include <hip/hip_bf16.h>

typedef __attribute__((ext_vector_type(16))) __bf16 v16bf;
typedef __attribute__((ext_vector_type(2)))  __bf16 v2bf;
typedef __attribute__((ext_vector_type(8)))  float  v8f;
typedef int v4i_vs __attribute__((vector_size(16)));   // matches builtin's V4i

#define HW    1024           // H*W
#define DHW   1048576        // D*H*W
#define PIX   32768          // N*H*W

#define AS1 __attribute__((address_space(1)))
#define AS3 __attribute__((address_space(3)))

#if defined(__AMDGCN__) && __has_builtin(__builtin_amdgcn_global_load_async_to_lds_b128)
#define ASYNC_LDS 1
#endif

// ---------------------------------------------------------------------------
// Fragment-layout helpers (CDNA5 wave32 WMMA bf16 16x16x32).
// A-frag (16x32, row=lane%16): lane half by bit3 of k; 16 bf16 contiguous/lane.
// B-frag (32x16): lanes 0-15 hold K=0..7,16..23, lanes 16-31 hold K=8..15,24..31.
// Both stored so each lane reads 32 contiguous bytes (2x ds_load_b128).
// frag element count = 512 bf16 = 1KB.
// ---------------------------------------------------------------------------

// Kernel A: build bf16, fragment-swizzled Wq^T, Kproj^T, Vproj in workspace.
__global__ void relaxq_prep(const float* __restrict__ codebook,
                            const float* __restrict__ wq,
                            const float* __restrict__ wk,
                            const float* __restrict__ wv,
                            __bf16* __restrict__ ws_wq,   // [M][16384]
                            __bf16* __restrict__ ws_kp,   // [M][32768]
                            __bf16* __restrict__ ws_vp) { // [M][32768]
  int t = blockIdx.x * 256 + threadIdx.x;
  if (t < 262144) {
    // kprojT element (m, c, k): dot(codebook[m,k,:], wk[m,c,:])
    int m = t >> 15, rem = t & 32767, c = rem >> 8, k = rem & 255;
    const float* cb = codebook + (size_t)(m * 256 + k) * 128;
    const float* wr = wk       + (size_t)(m * 128 + c) * 128;
    float acc = 0.f;
    #pragma unroll 8
    for (int d = 0; d < 128; ++d) acc += cb[d] * wr[d];
    int frag = (c >> 5) * 16 + (k >> 4);           // B-matrix rows=c(128), cols=k(256)
    int lane = ((c >> 3) & 1) * 16 + (k & 15);
    ws_kp[m * 32768 + frag * 512 + lane * 16 + (c & 7) + 8 * ((c >> 4) & 1)] = (__bf16)acc;
  } else if (t < 524288) {
    // vproj element (m, k, d): dot(codebook[m,k,:], wv[m,d,:])
    int t2 = t - 262144;
    int m = t2 >> 15, rem = t2 & 32767, k = rem >> 7, d = rem & 127;
    const float* cb = codebook + (size_t)(m * 256 + k) * 128;
    const float* wr = wv       + (size_t)(m * 128 + d) * 128;
    float acc = 0.f;
    #pragma unroll 8
    for (int e = 0; e < 128; ++e) acc += cb[e] * wr[e];
    int frag = (k >> 5) * 8 + (d >> 4);            // B-matrix rows=k(256), cols=d(128)
    int lane = ((k >> 3) & 1) * 16 + (d & 15);
    ws_vp[m * 32768 + frag * 512 + lane * 16 + (k & 7) + 8 * ((k >> 4) & 1)] = (__bf16)acc;
  } else if (t < 655360) {
    // wqT element: B[d][c] = wq[m,c,d]
    int t3 = t - 524288;
    int m = t3 >> 14, rem = t3 & 16383, d = rem >> 7, c = rem & 127;
    float v = wq[(size_t)(m * 128 + c) * 128 + d];
    int frag = (d >> 5) * 8 + (c >> 4);            // rows=d(128), cols=c(128)
    int lane = ((d >> 3) & 1) * 16 + (c & 15);
    ws_wq[m * 16384 + frag * 512 + lane * 16 + (d & 7) + 8 * ((d >> 4) & 1)] = (__bf16)v;
  }
}

__device__ __forceinline__ v16bf ld_frag(const __bf16* base, int frag, int lane) {
  return *(const v16bf*)(base + frag * 512 + lane * 16);
}

__device__ __forceinline__ v8f wmma_bf16(v16bf a, v16bf b, v8f c) {
  return __builtin_amdgcn_wmma_f32_16x16x32_bf16(false, a, false, b, (short)0, c,
                                                 false, false);
}

// ---------------------------------------------------------------------------
// Kernel B: fused  Q=X·WqT -> L=Q·KpT (+gumbel, online softmax, argmax)
//                  -> quantized = softmax·Vproj.  One block = 128 pixels x 1 group.
// ---------------------------------------------------------------------------
__launch_bounds__(256, 1)
__global__ void relaxq_main(const float* __restrict__ latent,
                            const float* __restrict__ unif,
                            const __bf16* __restrict__ ws_wq,
                            const __bf16* __restrict__ ws_kp,
                            const __bf16* __restrict__ ws_vp,
                            float* __restrict__ quant_out,
                            float* __restrict__ code_out,
                            float* __restrict__ logit_out) {
  __shared__ __align__(16) unsigned char smem[237568];
  __bf16* s_x  = (__bf16*)(smem);            // 32 KB  X tile, A-frag layout
  __bf16* s_q  = (__bf16*)(smem + 32768);    // 32 KB  Q tile, A-frag layout
  __bf16* s_wq = (__bf16*)(smem + 65536);    // 32 KB  WqT  B-frags
  __bf16* s_kp = (__bf16*)(smem + 98304);    // 64 KB  KprojT B-frags
  __bf16* s_vp = (__bf16*)(smem + 163840);   // 64 KB  Vproj  B-frags
  __bf16* s_s  = (__bf16*)(smem + 229376);   // 8 KB   per-wave S A-frag
  float*  s_o  = (float*)(smem + 98304);     // aliases kp+vp after main loop

  const int tid  = threadIdx.x;
  const int wave = tid >> 5, lane = tid & 31;
  const int hf = lane >> 4, colc = lane & 15;
  const int m = blockIdx.y;
  const int pbase = blockIdx.x * 128;        // 128-pixel tile, within one n
  const int n = pbase >> 10, hw0 = pbase & 1023;
  const long lat_base = (long)n * DHW + (long)(m * 128) * HW + hw0;

  // --- stage weights (already fragment-swizzled): async global->LDS DMA when
  //     available (no VGPR round-trip, tracked by ASYNCcnt), else b128 copies ---
#ifdef ASYNC_LDS
  {
    const uint4* gw = (const uint4*)(ws_wq + m * 16384); uint4* sw = (uint4*)s_wq;
    const uint4* gk = (const uint4*)(ws_kp + m * 32768); uint4* sk = (uint4*)s_kp;
    const uint4* gv = (const uint4*)(ws_vp + m * 32768); uint4* sv = (uint4*)s_vp;
    for (int i = tid; i < 2048; i += 256)
      __builtin_amdgcn_global_load_async_to_lds_b128(
          (AS1 v4i_vs*)(gw + i), (AS3 v4i_vs*)(sw + i), 0, 0);
    for (int i = tid; i < 4096; i += 256) {
      __builtin_amdgcn_global_load_async_to_lds_b128(
          (AS1 v4i_vs*)(gk + i), (AS3 v4i_vs*)(sk + i), 0, 0);
      __builtin_amdgcn_global_load_async_to_lds_b128(
          (AS1 v4i_vs*)(gv + i), (AS3 v4i_vs*)(sv + i), 0, 0);
    }
  }
#else
  { const uint4* g = (const uint4*)(ws_wq + m * 16384); uint4* s = (uint4*)s_wq;
    for (int i = tid; i < 2048; i += 256) s[i] = g[i]; }
  { const uint4* g = (const uint4*)(ws_kp + m * 32768); uint4* s = (uint4*)s_kp;
    for (int i = tid; i < 4096; i += 256) s[i] = g[i]; }
  { const uint4* g = (const uint4*)(ws_vp + m * 32768); uint4* s = (uint4*)s_vp;
    for (int i = tid; i < 4096; i += 256) s[i] = g[i]; }
#endif
  // --- stage latent tile: coalesced reads (128 consecutive pixels per channel),
  //     channel pairs share one A-frag dword -> single ds_store_b32 ---
  for (int i = tid; i < 8192; i += 256) {
    int cj = i >> 7, pl = i & 127;
    int c = cj * 2;
    float f0 = latent[lat_base + (long)c * HW + pl];
    float f1 = latent[lat_base + (long)(c + 1) * HW + pl];
    int kin  = c & 31;                      // even; pair shares ha/vv
    int frag = (pl >> 4) * 4 + (c >> 5);
    int ha   = (kin >> 3) & 1;
    int vv   = ((kin & 7) >> 1) + 4 * (kin >> 4);
    v2bf pr; pr.x = (__bf16)f0; pr.y = (__bf16)f1;
    *(v2bf*)(s_x + frag * 512 + (ha * 16 + (pl & 15)) * 16 + vv * 2) = pr;
  }
#ifdef ASYNC_LDS
#if __has_builtin(__builtin_amdgcn_s_wait_asynccnt)
  __builtin_amdgcn_s_wait_asynccnt(0);
#else
  asm volatile("s_wait_asynccnt 0" ::: "memory");
#endif
#endif
  __syncthreads();

  // --- GEMM1: Q[16x128] = X[16x128] . WqT, per wave (rows = its 16 pixels) ---
  v16bf xa[4];
  #pragma unroll
  for (int kc = 0; kc < 4; ++kc) xa[kc] = ld_frag(s_x, wave * 4 + kc, lane);

  #pragma unroll
  for (int ct = 0; ct < 8; ++ct) {
    v8f q = {0.f, 0.f, 0.f, 0.f, 0.f, 0.f, 0.f, 0.f};
    #pragma unroll
    for (int kc = 0; kc < 4; ++kc)
      q = wmma_bf16(xa[kc], ld_frag(s_wq, kc * 8 + ct, lane), q);
    // write Q (C/D layout) into per-wave A-frag region of s_q
    int c0 = ct * 16 + colc;
    int kin = c0 & 31;
    int frag = wave * 4 + (c0 >> 5);
    int ha = (kin >> 3) & 1;
    int vv = ((kin & 7) >> 1) + 4 * (kin >> 4);
    #pragma unroll
    for (int r = 0; r < 8; ++r)
      s_q[frag * 512 + (ha * 16 + r + 8 * hf) * 16 + vv * 2 + (kin & 1)] = (__bf16)q[r];
  }
  asm volatile("s_wait_dscnt 0" ::: "memory");   // wave-local LDS RAW
  v16bf aq[4];
  #pragma unroll
  for (int kc = 0; kc < 4; ++kc) aq[kc] = ld_frag(s_q, wave * 4 + kc, lane);

  // --- fused logits + gumbel + online softmax + quantize accumulation ---
  v8f acc[8];
  #pragma unroll
  for (int dt = 0; dt < 8; ++dt) acc[dt] = (v8f){0.f,0.f,0.f,0.f,0.f,0.f,0.f,0.f};
  float mold[8], ssum[8], amax[8]; int aidx[8];
  #pragma unroll
  for (int r = 0; r < 8; ++r) { mold[r] = -3e38f; ssum[r] = 0.f; amax[r] = -3e38f; aidx[r] = 0; }

  const int p0 = pbase + wave * 16;

  for (int kb = 0; kb < 256; kb += 32) {        // 8 chunks of 32 codes
    float lg[2][8];
    #pragma unroll
    for (int ct2 = 0; ct2 < 2; ++ct2) {
      v8f l = {0.f, 0.f, 0.f, 0.f, 0.f, 0.f, 0.f, 0.f};
      int ctile = (kb >> 4) + ct2;
      #pragma unroll
      for (int kc = 0; kc < 4; ++kc)
        l = wmma_bf16(aq[kc], ld_frag(s_kp, kc * 16 + ctile, lane), l);
      int k = kb + ct2 * 16 + colc;
      long gb = (long)(p0 + 8 * hf) * 2048 + m * 256 + k;
      #pragma unroll
      for (int r = 0; r < 8; ++r) {
        long gi = gb + (long)r * 2048;
        float lv = l[r];
        logit_out[gi] = lv;                      // raw logit output
        if (lv > amax[r]) { amax[r] = lv; aidx[r] = k; }
        float u = unif[gi];
        lg[ct2][r] = lv - __logf(-__logf(u));    // + gumbel
      }
    }
    // per-row (16-lane) online softmax update
    #pragma unroll
    for (int r = 0; r < 8; ++r) {
      float cm = fmaxf(lg[0][r], lg[1][r]);
      cm = fmaxf(cm, __shfl_xor(cm, 1, 32));
      cm = fmaxf(cm, __shfl_xor(cm, 2, 32));
      cm = fmaxf(cm, __shfl_xor(cm, 4, 32));
      cm = fmaxf(cm, __shfl_xor(cm, 8, 32));
      float mn = fmaxf(mold[r], cm);
      float sc = __expf(mold[r] - mn);
      float e0 = __expf(lg[0][r] - mn);
      float e1 = __expf(lg[1][r] - mn);
      float ps = e0 + e1;
      ps += __shfl_xor(ps, 1, 32);
      ps += __shfl_xor(ps, 2, 32);
      ps += __shfl_xor(ps, 4, 32);
      ps += __shfl_xor(ps, 8, 32);
      ssum[r] = ssum[r] * sc + ps;
      mold[r] = mn;
      #pragma unroll
      for (int dt = 0; dt < 8; ++dt) acc[dt][r] *= sc;    // flash rescale
      // stage S block (bf16) into this wave's A-fragment
      int row = r + 8 * hf;
      { int kl = colc;                                     // k-local 0..15
        s_s[wave * 512 + (((kl >> 3) & 1) * 16 + row) * 16 + ((kl & 7) >> 1) * 2 + (kl & 1)] = (__bf16)e0; }
      { int kl = 16 + colc;                                // k-local 16..31
        s_s[wave * 512 + (((kl >> 3) & 1) * 16 + row) * 16 + (((kl & 7) >> 1) + 4) * 2 + (kl & 1)] = (__bf16)e1; }
    }
    asm volatile("s_wait_dscnt 0" ::: "memory");
    v16bf sa = *(const v16bf*)(s_s + wave * 512 + lane * 16);
    int kc3 = kb >> 5;
    #pragma unroll
    for (int dt = 0; dt < 8; ++dt)                         // GEMM3 accumulate
      acc[dt] = wmma_bf16(sa, ld_frag(s_vp, kc3 * 8 + dt, lane), acc[dt]);
  }

  // --- argmax -> code (float-encoded uint8) ---
  #pragma unroll
  for (int r = 0; r < 8; ++r) {
    float am = amax[r]; int ai = aidx[r];
    #pragma unroll
    for (int off = 1; off < 16; off <<= 1) {
      float om = __shfl_xor(am, off, 32);
      int   oi = __shfl_xor(ai, off, 32);
      if (om > am || (om == am && oi < ai)) { am = om; ai = oi; }
    }
    if (colc == r) code_out[(long)(p0 + r + 8 * hf) * 8 + m] = (float)ai;
  }

  // --- normalize + stage quantized through LDS for coalesced NCHW store ---
  __syncthreads();                        // all kp/vp reads done; alias as s_o
  float rs[8];
  #pragma unroll
  for (int r = 0; r < 8; ++r) rs[r] = 1.f / ssum[r];
  #pragma unroll
  for (int dt = 0; dt < 8; ++dt) {
    int d = dt * 16 + colc;
    #pragma unroll
    for (int r = 0; r < 8; ++r)
      s_o[d * 133 + wave * 16 + r + 8 * hf] = acc[dt][r] * rs[r];
  }
  __syncthreads();
  for (int i = tid; i < 16384; i += 256) {
    int d = i >> 7, pl = i & 127;
    quant_out[(long)n * DHW + (long)(m * 128 + d) * HW + hw0 + pl] = s_o[d * 133 + pl];
  }
}

// ---------------------------------------------------------------------------
extern "C" void kernel_launch(void* const* d_in, const int* in_sizes, int n_in,
                              void* d_out, int out_size, void* d_ws, size_t ws_size,
                              hipStream_t stream) {
  const float* latent   = (const float*)d_in[0];
  const float* codebook = (const float*)d_in[1];
  const float* wq       = (const float*)d_in[2];
  const float* wk       = (const float*)d_in[3];
  const float* wv       = (const float*)d_in[4];
  const float* unif     = (const float*)d_in[5];

  float* out   = (float*)d_out;
  float* quant = out;                    // [1,32,1024,32,32] = 33,554,432
  float* code  = out + 33554432;         // [32,32,32,8]      =    262,144
  float* logit = out + 33816576;         // [32,32,32,8,256]  = 67,108,864

  __bf16* ws_wq = (__bf16*)d_ws;                       // 8*16384 bf16
  __bf16* ws_kp = (__bf16*)d_ws + 131072;              // 8*32768 bf16
  __bf16* ws_vp = (__bf16*)d_ws + 131072 + 262144;     // 8*32768 bf16

  relaxq_prep<<<2560, 256, 0, stream>>>(codebook, wq, wk, wv, ws_wq, ws_kp, ws_vp);

  dim3 grid(256, 8);   // 256 pixel tiles x 8 groups
  relaxq_main<<<grid, 256, 0, stream>>>(latent, unif, ws_wq, ws_kp, ws_vp,
                                        quant, code, logit);
}
